// LSTM_83167746720239
// MI455X (gfx1250) — matrix operand order repaired and checked by reading, not compiled, not run on hardware
//
#include <hip/hip_runtime.h>

// ---------------- problem dims ----------------
#define TSEQ   2048
#define NWG    16      // persistent workgroups; WG w owns h columns [16w,16w+16)
#define NTHR   256     // 8 waves; wave v owns batch M-tile v (rows 16v..16v+15)

typedef __attribute__((ext_vector_type(16))) __bf16 v16bf;
typedef __attribute__((ext_vector_type(8)))  float  v8f;

// ---------------- workspace layout (bytes) ----------------
// x_sw : bf16 [T][kc=4][m=8][lane=32][e=16]   (A-fragment swizzled x)
#define OFF_XSW   0ull                  // 67,108,864 B
// w_sw : bf16 [w=16][g=4][kc=12][lane=32][e=16] (B-fragments of [W_ih|W_hh])
#define OFF_WSW   67108864ull           //    786,432 B
// wo_sw: bf16 [nt=4][kc=8][lane=32][e=16]     (B-fragments of W_out)
#define OFF_WOSW  67895296ull           //     32,768 B
// h_sw : bf16 2 x [kc=8][m=8][lane=32][e=16]  (ping-pong hidden state)
#define OFF_HSW   67928064ull           //    131,072 B
#define OFF_BIAS  68059136ull           //      4,096 B (b_ih + b_hh, f32)
#define OFF_SYNC  68063232ull           //  u32 count, u32 generation

__device__ __forceinline__ float sigm(float x)   { return 1.0f / (1.0f + __expf(-x)); }
__device__ __forceinline__ float tanh_f(float x) { return 2.0f / (1.0f + __expf(-2.0f * x)) - 1.0f; }

// A-fragment element->K map (16-bit A 16x32, ISA 7.12.2):
// half=lane/16 :  e<8 -> k = e + 8*half ;  e>=8 -> k = e + 8 + 8*half
__device__ __forceinline__ unsigned a_k_of_e(unsigned e, unsigned half) {
  return e + ((e >= 8u) ? 8u : 0u) + half * 8u;
}

// ---------------- prep: swizzle x into bf16 A-fragments ----------------
__global__ __launch_bounds__(256) void k_prep_x(const float* __restrict__ x, void* xswv) {
  __bf16* xsw = (__bf16*)xswv;
  size_t i = (size_t)blockIdx.x * blockDim.x + threadIdx.x;
  if (i >= (size_t)TSEQ * 16384u) return;
  unsigned e  = i & 15u;
  unsigned L  = (i >> 4) & 31u;
  unsigned m  = (i >> 9) & 7u;
  unsigned kc = (i >> 12) & 3u;
  unsigned t  = (unsigned)(i >> 14);
  unsigned k   = a_k_of_e(e, L >> 4);
  unsigned row = m * 16u + (L & 15u);
  unsigned col = kc * 32u + k;
  xsw[i] = (__bf16)x[((size_t)t * 128u + row) * 128u + col];
}

// ---------------- prep: B-fragments of W_cat = [W_ih | W_hh] ----------------
// B 32x16 layout: same K order as A: e<8 -> k = e + 8*half ; e>=8 -> k = e + 8 + 8*half
__global__ __launch_bounds__(256) void k_prep_w(const float* __restrict__ Wih,
                                                const float* __restrict__ Whh, void* wswv) {
  __bf16* wsw = (__bf16*)wswv;
  unsigned i = blockIdx.x * blockDim.x + threadIdx.x;
  if (i >= 393216u) return;
  unsigned e = i & 15u, L = (i >> 4) & 31u;
  unsigned within = i >> 9;
  unsigned kc = within % 12u, g = (within / 12u) & 3u, w = within / 48u;
  unsigned k    = kc * 32u + a_k_of_e(e, L >> 4);
  unsigned gcol = g * 256u + w * 16u + (L & 15u);
  float v = (k < 128u) ? Wih[(size_t)gcol * 128u + k] : Whh[(size_t)gcol * 256u + (k - 128u)];
  wsw[i] = (__bf16)v;
}

__global__ __launch_bounds__(256) void k_prep_wout(const float* __restrict__ Wout, void* wov) {
  __bf16* wo = (__bf16*)wov;
  unsigned i = blockIdx.x * blockDim.x + threadIdx.x;
  if (i >= 16384u) return;
  unsigned e = i & 15u, L = (i >> 4) & 31u;
  unsigned within = i >> 9;
  unsigned kc = within & 7u, nt = within >> 3;
  unsigned k    = kc * 32u + a_k_of_e(e, L >> 4);
  unsigned ocol = nt * 16u + (L & 15u);
  wo[i] = (__bf16)Wout[(size_t)ocol * 256u + k];
}

// zero h ping-pong, combine biases, reset barrier (every launch -> graph-safe)
__global__ __launch_bounds__(256) void k_prep_misc(const float* __restrict__ bih,
                                                   const float* __restrict__ bhh,
                                                   void* hswv, float* bias, unsigned* sync) {
  unsigned i = blockIdx.x * blockDim.x + threadIdx.x;
  if (i < 65536u)       ((__bf16*)hswv)[i] = (__bf16)0.0f;
  else if (i < 66560u)  { unsigned j = i - 65536u; bias[j] = bih[j] + bhh[j]; }
  else if (i < 66562u)  sync[i - 66560u] = 0u;
}

// ---------------- split grid barrier (16 resident WGs) ----------------
// arrive: non-blocking; last arriver resets count and bumps generation (release)
__device__ __forceinline__ void barrier_arrive(unsigned* cnt, unsigned* gen, unsigned nwg) {
  __threadfence();          // make this thread's h stores device-visible
  __syncthreads();          // all waves of the WG have flushed
  if (threadIdx.x == 0) {
    if (__hip_atomic_fetch_add(cnt, 1u, __ATOMIC_ACQ_REL, __HIP_MEMORY_SCOPE_AGENT) == nwg - 1u) {
      __hip_atomic_store(cnt, 0u, __ATOMIC_RELAXED, __HIP_MEMORY_SCOPE_AGENT);
      __hip_atomic_fetch_add(gen, 1u, __ATOMIC_RELEASE, __HIP_MEMORY_SCOPE_AGENT);
    }
  }
}
// wait: spin until generation reaches target (acquire -> invalidates WGP$)
__device__ __forceinline__ void barrier_wait(unsigned* gen, unsigned target) {
  if (threadIdx.x == 0) {
    while (__hip_atomic_load(gen, __ATOMIC_ACQUIRE, __HIP_MEMORY_SCOPE_AGENT) < target)
      __builtin_amdgcn_s_sleep(1);
  }
  __syncthreads();
}

// fused output projection out[tt] = h_tt @ W_out^T + b_out  (one M-tile, two N-tiles)
__device__ __forceinline__ void do_out(int tt, const __bf16* hb, const __bf16* wo,
                                       int mo, int nb, int lane, int col, int half_d,
                                       const float* bo, float* __restrict__ out) {
  v8f oa[2];
  #pragma unroll
  for (int nt = 0; nt < 2; ++nt)
    #pragma unroll
    for (int j = 0; j < 8; ++j) oa[nt][j] = bo[nt];
  #pragma unroll
  for (int kc = 0; kc < 8; ++kc) {
    v16bf a = *(const v16bf*)(hb + (kc * 8 + mo) * 512 + lane * 16);
    #pragma unroll
    for (int nt = 0; nt < 2; ++nt) {
      v16bf b = *(const v16bf*)(wo + ((size_t)(nb + nt) * 8 + kc) * 512 + lane * 16);
      oa[nt] = __builtin_amdgcn_wmma_f32_16x16x32_bf16(false, a, false, b,
                                                       (short)0, oa[nt], false, false);
    }
  }
  #pragma unroll
  for (int nt = 0; nt < 2; ++nt)
    #pragma unroll
    for (int j = 0; j < 8; ++j)
      out[((size_t)tt * 128 + mo * 16 + j + 8 * half_d) * 64 + (nb + nt) * 16 + col] = oa[nt][j];
}

// ---------------- persistent fused LSTM ----------------
__global__ __launch_bounds__(NTHR, 1)
void k_lstm(const void* xswv, const void* wswv, const void* wov, const float* __restrict__ bias,
            void* hswv, unsigned* sync, const float* __restrict__ b_out, float* __restrict__ out) {
  const int w      = blockIdx.x;        // H-slice / gate-column slice owner
  const int tid    = threadIdx.x;
  const int wv     = tid >> 5;          // wave = batch M-tile
  const int lane   = tid & 31;
  const int col    = lane & 15;         // N column within 16x16 tile
  const int half_d = lane >> 4;         // C/D: row = j + 8*half_d

  // weights for this WG's 64 gate columns live in LDS for the whole kernel
  __shared__ __bf16 s_w[4 * 12 * 512]; // [gate][kc][lane*16+e]  48 KB
  __shared__ float  s_bias[64];

  {
    const uint4* src = (const uint4*)((const __bf16*)wswv + (size_t)w * 24576u);
    uint4* dst = (uint4*)s_w;
    #pragma unroll
    for (int i = tid; i < 3072; i += NTHR) dst[i] = src[i];
    if (tid < 64) s_bias[tid] = bias[(tid >> 4) * 256 + w * 16 + (tid & 15)];
  }
  __syncthreads();

  const __bf16* xsw = (const __bf16*)xswv;
  const __bf16* wo  = (const __bf16*)wov;
  __bf16*       hsw = (__bf16*)hswv;
  unsigned* cnt = &sync[0];
  unsigned* gen = &sync[1];

  // cell state lives in VGPRs (this WG owns exactly these 16 h/c columns)
  float cc[8];
  #pragma unroll
  for (int j = 0; j < 8; ++j) cc[j] = 0.0f;

  // bias pre-splatted into accumulator seeds
  v8f bacc[4];
  #pragma unroll
  for (int g = 0; g < 4; ++g) {
    float bv = s_bias[g * 16 + col];
    #pragma unroll
    for (int j = 0; j < 8; ++j) bacc[g][j] = bv;
  }

  // h scatter target: write h in A-fragment layout for next step's loads.
  // kglob = 16w+col; kk = kglob&31 -> (half_a, e) inverse of a_k_of_e.
  const int kglob = w * 16 + col;
  const int kch   = kglob >> 5;
  const int kk    = kglob & 31;
  int ha, ee;
  if (kk < 8)       { ha = 0; ee = kk; }
  else if (kk < 16) { ha = 1; ee = kk - 8; }
  else if (kk < 24) { ha = 0; ee = kk - 8; }
  else              { ha = 1; ee = kk - 16; }
  const int hdest_base = (kch * 8 + wv) * 512 + (ha * 16 + 8 * half_d) * 16 + ee; // + j*16

  // output-projection tile assignment: WG w -> M-tile w&7, N-tiles {2*(w>>3), +1}
  const int mo = w & 7;
  const int nb = (w >> 3) << 1;
  float bo[2] = { b_out[nb * 16 + col], b_out[(nb + 1) * 16 + col] };

  for (int t = 0; t < TSEQ; ++t) {
    const int cur = t & 1;                      // holds h_{t-1} (once barrier passed)
    const __bf16* hc = hsw + cur * 32768;
    __bf16*       hn = hsw + (cur ^ 1) * 32768;

    v8f acc[4] = { bacc[0], bacc[1], bacc[2], bacc[3] };

    if (t + 1 < TSEQ)  // pull next timestep's x fragments toward L2/L0
      __builtin_prefetch((const void*)(xsw + (((size_t)(t + 1) * 4) * 8 + wv) * 512 + lane * 16), 0, 3);

    // --- independent work BEFORE the barrier wait ---
    // K = 0..127 : input contribution (x_t), A-fragments pre-swizzled in global
    #pragma unroll
    for (int kc = 0; kc < 4; ++kc) {
      v16bf a = *(const v16bf*)(xsw + (((size_t)t * 4 + kc) * 8 + wv) * 512 + lane * 16);
      #pragma unroll
      for (int g = 0; g < 4; ++g) {
        v16bf b = *(const v16bf*)(&s_w[(g * 12 + kc) * 512 + lane * 16]);
        acc[g] = __builtin_amdgcn_wmma_f32_16x16x32_bf16(false, a, false, b,
                                                         (short)0, acc[g], false, false);
      }
    }

    // --- wait for h_{t-1} from all WGs (generation t) ---
    if (t > 0) {
      barrier_wait(gen, (unsigned)t);
      // h_{t-1} now globally visible: wave 0 emits out[t-1] while others run the h-part
      if (wv == 0) do_out(t - 1, hc, wo, mo, nb, lane, col, half_d, bo, out);
    }

    // K = 128..383 : recurrent contribution (h_{t-1})
    #pragma unroll
    for (int kc = 0; kc < 8; ++kc) {
      v16bf a = *(const v16bf*)(hc + (kc * 8 + wv) * 512 + lane * 16);
      #pragma unroll
      for (int g = 0; g < 4; ++g) {
        v16bf b = *(const v16bf*)(&s_w[(g * 12 + 4 + kc) * 512 + lane * 16]);
        acc[g] = __builtin_amdgcn_wmma_f32_16x16x32_bf16(false, a, false, b,
                                                         (short)0, acc[g], false, false);
      }
    }

    // gate activations, cell update, h written directly in A-fragment layout
    #pragma unroll
    for (int j = 0; j < 8; ++j) {
      float iv = sigm(acc[0][j]);
      float fv = sigm(acc[1][j]);
      float gv = tanh_f(acc[2][j]);
      float ov = sigm(acc[3][j]);
      float cn = fv * cc[j] + iv * gv;
      cc[j] = cn;
      hn[hdest_base + j * 16] = (__bf16)(ov * tanh_f(cn));
    }

    barrier_arrive(cnt, gen, gridDim.x);        // non-blocking; bumps gen -> t+1
  }

  // epilogue: out[T-1] once the final h is globally visible
  barrier_wait(gen, (unsigned)TSEQ);
  if (wv == 0) do_out(TSEQ - 1, hsw + (TSEQ & 1) * 32768, wo, mo, nb, lane, col, half_d, bo, out);
}

// ---------------- launch ----------------
extern "C" void kernel_launch(void* const* d_in, const int* in_sizes, int n_in,
                              void* d_out, int out_size, void* d_ws, size_t ws_size,
                              hipStream_t stream) {
  (void)in_sizes; (void)n_in; (void)out_size; (void)ws_size;
  const float* x    = (const float*)d_in[0];
  const float* Wih  = (const float*)d_in[1];
  const float* Whh  = (const float*)d_in[2];
  const float* bih  = (const float*)d_in[3];
  const float* bhh  = (const float*)d_in[4];
  const float* Wout = (const float*)d_in[5];
  const float* bout = (const float*)d_in[6];

  char* ws = (char*)d_ws;
  void*     xsw  = (void*)(ws + OFF_XSW);
  void*     wsw  = (void*)(ws + OFF_WSW);
  void*     wosw = (void*)(ws + OFF_WOSW);
  void*     hsw  = (void*)(ws + OFF_HSW);
  float*    bias = (float*)(ws + OFF_BIAS);
  unsigned* sync = (unsigned*)(ws + OFF_SYNC);

  k_prep_x   <<<131072, 256, 0, stream>>>(x, xsw);
  k_prep_w   <<<1536,   256, 0, stream>>>(Wih, Whh, wsw);
  k_prep_wout<<<64,     256, 0, stream>>>(Wout, wosw);
  k_prep_misc<<<261,    256, 0, stream>>>(bih, bhh, hsw, bias, sync);
  k_lstm     <<<NWG,   NTHR, 0, stream>>>(xsw, wsw, wosw, bias, hsw, sync, bout, (float*)d_out);
}
